// CausalLiquidCell_30528627540772
// MI455X (gfx1250) — matrix operand order repaired and checked by reading, not compiled, not run on hardware
//
#include <hip/hip_runtime.h>
#include <stdint.h>

typedef __attribute__((ext_vector_type(16))) __bf16          v16bf;
typedef __attribute__((ext_vector_type(16))) unsigned short  v16u;
typedef __attribute__((ext_vector_type(8)))  float           v8f;

union FragQ { v16u v; uint4 q[2]; };

__device__ __forceinline__ unsigned short f2bf(float f) {
  unsigned u = __float_as_uint(f);
  return (unsigned short)((u + 0x7FFFu + ((u >> 16) & 1u)) >> 16);  // RNE
}

__device__ __forceinline__ float dot4(float4 a, float4 b) {
  return a.x * b.x + a.y * b.y + a.z * b.z + a.w * b.w;
}

__device__ __forceinline__ v8f wmma_bf16(v16u a, v16u b, v8f c) {
  return __builtin_amdgcn_wmma_f32_16x16x32_bf16(
      false, __builtin_bit_cast(v16bf, a),
      false, __builtin_bit_cast(v16bf, b),
      (short)0, c, false, false);
}

// ---- prologue: fp32 weights -> bf16 scratch: [Win 32768][attn 65536][rec 65536]
__global__ void cvt_weights(const float* __restrict__ win,
                            const float* __restrict__ attnw,
                            const float* __restrict__ recw,
                            unsigned short* __restrict__ ws) {
  int i = blockIdx.x * blockDim.x + threadIdx.x;   // 0..65535
  if (i < 32768) ws[i] = f2bf(win[i]);
  ws[32768 + i] = f2bf(attnw[i]);
  ws[98304 + i] = f2bf(recw[i]);
}

#define LDS_ATTN 0
#define LDS_REC  131072
#define LDS_EV   262144

__global__ void __launch_bounds__(256)
liquid_main(const float* __restrict__ x, const float* __restrict__ prev,
            const float* __restrict__ hidden,
            const float* __restrict__ Winb, const float* __restrict__ Wrecb,
            const float* __restrict__ attnb,
            const float* __restrict__ evwv, const float* __restrict__ evb,
            const float* __restrict__ tau,
            const unsigned short* __restrict__ wsbf,
            float* __restrict__ out, int B) {
  extern __shared__ char smem[];
  const int tid  = threadIdx.x;
  const int wid  = tid >> 5;
  const int lane = tid & 31;
  const int h    = lane >> 4;   // half-wave select
  const int ml   = lane & 15;

  // ---- stage attn_w + W_rec_w (bf16, 256KB) into LDS with async copies ----
  {
    const char* src = (const char*)wsbf + 65536;            // attn..rec region
    unsigned lbase = (unsigned)(size_t)(&smem[0]);
#pragma unroll 4
    for (int it = 0; it < 64; ++it) {
      int idx = it * 4096 + tid * 16;
      unsigned la = lbase + (unsigned)idx;
      unsigned long long ga = (unsigned long long)(size_t)(src + idx);
      asm volatile("global_load_async_to_lds_b128 %0, %1, off"
                   :: "v"(la), "v"(ga) : "memory");
    }
    asm volatile("s_wait_asynccnt 0" ::: "memory");
    __syncthreads();
  }

  const float ev_bias = evb[0];
  float* evbuf = (float*)(smem + LDS_EV) + wid * 16;
  const int nTiles = B >> 4;
  const int wpb    = blockDim.x >> 5;
  const int totW   = gridDim.x * wpb;

  for (int t = blockIdx.x * wpb + wid; t < nTiles; t += totW) {
    const int m0  = t << 4;
    const int row = m0 + ml;
    const float* xr = x + (size_t)row * 128;
    const float* pr = prev + (size_t)row * 128;
    const float* hr = hidden + (size_t)row * 256;

    if (t + totW < nTiles) {
      __builtin_prefetch(x + ((size_t)(t + totW) * 16 + ml) * 128, 0, 0);
      __builtin_prefetch(hidden + ((size_t)(t + totW) * 16 + ml) * 256, 0, 0);
    }

    // ---- phase 1: x A-frags (K=128), event-weight dot, hidden^T B-frags ----
    v16u xA[4];
    float s = 0.f;
#pragma unroll
    for (int f = 0; f < 4; ++f) {
      const int k1 = 32 * f + 16 * h, k2 = k1 + 8;
      float4 a0 = *(const float4*)(xr + k1);
      float4 a1 = *(const float4*)(xr + k1 + 4);
      float4 a2 = *(const float4*)(xr + k2);
      float4 a3 = *(const float4*)(xr + k2 + 4);
      float4 w0 = *(const float4*)(evwv + k1);
      float4 w1 = *(const float4*)(evwv + k1 + 4);
      float4 w2 = *(const float4*)(evwv + k2);
      float4 w3 = *(const float4*)(evwv + k2 + 4);
      float4 p0 = *(const float4*)(pr + k1);
      float4 p1 = *(const float4*)(pr + k1 + 4);
      float4 p2 = *(const float4*)(pr + k2);
      float4 p3 = *(const float4*)(pr + k2 + 4);
      float4 u0 = *(const float4*)(evwv + 128 + k1);
      float4 u1 = *(const float4*)(evwv + 128 + k1 + 4);
      float4 u2 = *(const float4*)(evwv + 128 + k2);
      float4 u3 = *(const float4*)(evwv + 128 + k2 + 4);
      s += dot4(a0, w0) + dot4(a1, w1) + dot4(a2, w2) + dot4(a3, w3);
      s += dot4(p0, u0) + dot4(p1, u1) + dot4(p2, u2) + dot4(p3, u3);
      v16u v;
      v[0]=f2bf(a0.x);  v[1]=f2bf(a0.y);  v[2]=f2bf(a0.z);  v[3]=f2bf(a0.w);
      v[4]=f2bf(a1.x);  v[5]=f2bf(a1.y);  v[6]=f2bf(a1.z);  v[7]=f2bf(a1.w);
      v[8]=f2bf(a2.x);  v[9]=f2bf(a2.y);  v[10]=f2bf(a2.z); v[11]=f2bf(a2.w);
      v[12]=f2bf(a3.x); v[13]=f2bf(a3.y); v[14]=f2bf(a3.z); v[15]=f2bf(a3.w);
      xA[f] = v;
    }
    s += __shfl_xor(s, 16);
    const float evw = 1.f / (1.f + __expf(-(s + ev_bias)));
    if (h == 0) {
      out[(size_t)B * 256 + row] = evw;   // event_weight output
      evbuf[ml] = 1.f + evw;
    }

    v16u hB[8];
#pragma unroll
    for (int f = 0; f < 8; ++f) {
      const int k = 32 * f + 16 * h;
      float4 c0 = *(const float4*)(hr + k);
      float4 c1 = *(const float4*)(hr + k + 4);
      float4 c2 = *(const float4*)(hr + k + 8);
      float4 c3 = *(const float4*)(hr + k + 12);
      v16u v;
      v[0]=f2bf(c0.x);  v[1]=f2bf(c0.y);  v[2]=f2bf(c0.z);  v[3]=f2bf(c0.w);
      v[4]=f2bf(c1.x);  v[5]=f2bf(c1.y);  v[6]=f2bf(c1.z);  v[7]=f2bf(c1.w);
      v[8]=f2bf(c2.x);  v[9]=f2bf(c2.y);  v[10]=f2bf(c2.z); v[11]=f2bf(c2.w);
      v[12]=f2bf(c3.x); v[13]=f2bf(c3.y); v[14]=f2bf(c3.z); v[15]=f2bf(c3.w);
      hB[f] = v;
    }

    // ---- phase 2: attn^T = attn_w x hidden^T; gate -> rec-GEMM A-frags ----
    v16u gA[8];
#pragma unroll
    for (int tn = 0; tn < 16; ++tn) {
      v8f acc = {0.f,0.f,0.f,0.f,0.f,0.f,0.f,0.f};
      const int wrow = tn * 16 + ml;
#pragma unroll
      for (int f = 0; f < 8; ++f) {
        const int byt = LDS_ATTN + (wrow * 256 + 32 * f + 16 * h) * 2;
        FragQ q;
        q.q[0] = *(const uint4*)(smem + byt);
        q.q[1] = *(const uint4*)(smem + byt + 16);
        acc = wmma_bf16(q.v, hB[f], acc);
      }
      float4 b0 = *(const float4*)(attnb + tn * 16 + 8 * h);
      float4 b1 = *(const float4*)(attnb + tn * 16 + 8 * h + 4);
      float4 h0 = *(const float4*)(hr + tn * 16 + 8 * h);
      float4 h1 = *(const float4*)(hr + tn * 16 + 8 * h + 4);
      const float bb[8] = {b0.x,b0.y,b0.z,b0.w,b1.x,b1.y,b1.z,b1.w};
      const float hh[8] = {h0.x,h0.y,h0.z,h0.w,h1.x,h1.y,h1.z,h1.w};
#pragma unroll
      for (int v = 0; v < 8; ++v) {
        float aw = 1.f / (1.f + __expf(-(acc[v] + bb[v])));
        gA[tn >> 1][(tn & 1) * 8 + v] = f2bf(hh[v] * aw);
      }
    }

    // ---- phase 3: input + recurrent GEMMs, pointwise liquid update ----
    float ev8[8];
    {
      const float4 e0 = *(const float4*)(evbuf + 8 * h);
      const float4 e1 = *(const float4*)(evbuf + 8 * h + 4);
      ev8[0]=e0.x; ev8[1]=e0.y; ev8[2]=e0.z; ev8[3]=e0.w;
      ev8[4]=e1.x; ev8[5]=e1.y; ev8[6]=e1.z; ev8[7]=e1.w;
    }
    for (int tn = 0; tn < 16; ++tn) {
      const int n = tn * 16 + ml;
      v8f acc1 = {0.f,0.f,0.f,0.f,0.f,0.f,0.f,0.f};
      v8f acc2 = {0.f,0.f,0.f,0.f,0.f,0.f,0.f,0.f};
      const unsigned short* wb = wsbf + n * 128;   // W_in row (bf16, L2)
#pragma unroll
      for (int f = 0; f < 4; ++f) {
        const int off = 32 * f + 16 * h;
        FragQ q;
        q.q[0] = *(const uint4*)(wb + off);
        q.q[1] = *(const uint4*)(wb + off + 8);
        acc1 = wmma_bf16(xA[f], q.v, acc1);
      }
#pragma unroll
      for (int f = 0; f < 8; ++f) {
        const int byt = LDS_REC + (n * 256 + 32 * f + 8 * h) * 2;
        FragQ q;
        q.q[0] = *(const uint4*)(smem + byt);
        q.q[1] = *(const uint4*)(smem + byt + 32);
        acc2 = wmma_bf16(gA[f], q.v, acc2);
      }
      const float bin = Winb[n];
      const float brc = Wrecb[n];
      const float tc  = fminf(fmaxf(tau[n], 0.1f), 10.f);
#pragma unroll
      for (int v = 0; v < 8; ++v) {
        const int m = m0 + v + 8 * h;
        const float hv = hidden[(size_t)m * 256 + n];
        const float t1 = tanhf(acc1[v] + bin);
        const float t2 = tanhf(acc2[v] + brc);
        out[(size_t)m * 256 + n] = hv + 0.1f * ((-hv + t1 + t2) / tc) * ev8[v];
      }
    }
  }
}

extern "C" void kernel_launch(void* const* d_in, const int* in_sizes, int n_in,
                              void* d_out, int out_size, void* d_ws, size_t ws_size,
                              hipStream_t stream) {
  const float* x     = (const float*)d_in[0];
  const float* prev  = (const float*)d_in[1];
  const float* hid   = (const float*)d_in[2];
  const float* winw  = (const float*)d_in[3];
  const float* winb  = (const float*)d_in[4];
  const float* wrecw = (const float*)d_in[5];
  const float* wrecb = (const float*)d_in[6];
  const float* attnw = (const float*)d_in[7];
  const float* attnb = (const float*)d_in[8];
  const float* evw   = (const float*)d_in[9];
  const float* evb   = (const float*)d_in[10];
  const float* tau   = (const float*)d_in[11];
  float* out = (float*)d_out;
  unsigned short* ws = (unsigned short*)d_ws;
  const int B = in_sizes[0] / 128;

  cvt_weights<<<256, 256, 0, stream>>>(winw, attnw, wrecw, ws);
  // 512 WGs x 8 waves -> 4096 wave-slots, one 16-row tile per wave (grid-stride safe)
  liquid_main<<<512, 256, 262656, stream>>>(x, prev, hid, winb, wrecb, attnb,
                                            evw, evb, tau, ws, out, B);
}
